// ReadingNetwork_62508954026162
// MI455X (gfx1250) — matrix operand order repaired and checked by reading, not compiled, not run on hardware
//
#include <hip/hip_runtime.h>
#include <hip/hip_bf16.h>

// ---------------------------------------------------------------------------
// ReadingNetwork on MI455X (gfx1250, wave32, WMMA bf16)
//   B=4096, NINP=NOUT=1024, NSLOTS=15
// Bandwidth-bound problem (~1 GB moved -> ~45us floor @ 23.3 TB/s).
// GEMMs run in bf16 WMMA (v_wmma_f32_16x16x32_bf16); tile staging uses the
// gfx1250 async Global->LDS path (ASYNCcnt); the memory_h/c sweep is fused
// with the rolling-shift update.
// ---------------------------------------------------------------------------

#define B_      4096
#define NINP_   1024
#define NOUT_   1024
#define NSLOTS_ 15
#define KCAT_   2048   // NINP + NOUT

typedef __bf16 bf16_t;
typedef __attribute__((ext_vector_type(16))) __bf16 v16bf;
typedef __attribute__((ext_vector_type(8)))  __bf16 v8bf;
typedef __attribute__((ext_vector_type(8)))  float  v8f;
typedef int v4i_ __attribute__((__vector_size__(16)));   // matches builtin param type

#if __has_builtin(__builtin_amdgcn_global_load_async_to_lds_b128) && \
    __has_builtin(__builtin_amdgcn_s_wait_asynccnt)
#define HAVE_ASYNC_LDS 1
// IOFFSET is added to BOTH the global and the LDS address (ISA 08 §4.4),
// so one base pointer pair + two immediate offsets stages 32B per lane.
#define ASYNC_CP16(gptr, lptr, off)                                       \
    __builtin_amdgcn_global_load_async_to_lds_b128(                       \
        (__attribute__((address_space(1))) v4i_*)(gptr),                  \
        (__attribute__((address_space(3))) v4i_*)(lptr), (off), 0)
#else
#define HAVE_ASYNC_LDS 0
#endif

static __device__ __forceinline__ v16bf join8(v8bf a, v8bf b) {
    return __builtin_shufflevector(a, b, 0,1,2,3,4,5,6,7,8,9,10,11,12,13,14,15);
}
static __device__ __forceinline__ float sigmoidf_(float x) {
    return 1.0f / (1.0f + __expf(-x));
}

// ---------------------------------------------------------------------------
// Pack kernels: f32 -> bf16 staging into workspace
// ---------------------------------------------------------------------------
__global__ __launch_bounds__(256) void pack_a_kernel(
    const float* __restrict__ input, const float* __restrict__ memory_h,
    bf16_t* __restrict__ Xcat, bf16_t* __restrict__ A2)
{
    size_t i = (size_t)blockIdx.x * 256 + threadIdx.x;   // < B*NINP
    size_t b = i >> 10;
    size_t d = i & 1023;
    float x = input[i];
    Xcat[b * KCAT_ + d]         = (bf16_t)x;
    Xcat[b * KCAT_ + NINP_ + d] = (bf16_t)memory_h[b * (size_t)NSLOTS_ * NOUT_ + d]; // slot 0
    A2[b * KCAT_ + d]           = (bf16_t)x;
}

__global__ __launch_bounds__(256) void pack_wp_kernel(
    const float* __restrict__ Wp, bf16_t* __restrict__ WpB)
{
    size_t i = (size_t)blockIdx.x * 256 + threadIdx.x;   // < NOUT*KCAT
    WpB[i] = (bf16_t)Wp[i];
}

__global__ __launch_bounds__(256) void pack_w2_kernel(
    const float* __restrict__ Wih, const float* __restrict__ Whh,
    bf16_t* __restrict__ W2)
{
    size_t i = (size_t)blockIdx.x * 256 + threadIdx.x;   // < 4*NOUT*KCAT
    size_t k  = i & 2047;
    size_t gr = i >> 11;                                 // g*1024 + n  (gate-major row)
    float v = (k < NINP_) ? Wih[gr * NINP_ + k]
                          : Whh[gr * NOUT_ + (k - NINP_)];
    W2[i] = (bf16_t)v;
}

__global__ __launch_bounds__(256) void pack_bias_kernel(
    const float* __restrict__ b_ih, const float* __restrict__ b_hh,
    float* __restrict__ bias2)
{
    size_t i = (size_t)blockIdx.x * 256 + threadIdx.x;   // < 4*NOUT
    bias2[i] = b_ih[i] + b_hh[i];
}

// ---------------------------------------------------------------------------
// GEMM1: key[B,1024] = Xcat[B,2048] @ WpB[1024,2048]^T + b_proj
// Block tile 128x64, BK=32, 256 threads (8 waves, 4x2), wave tile 32x32
// ---------------------------------------------------------------------------
__global__ __launch_bounds__(256) void gemm_key_kernel(
    const bf16_t* __restrict__ Xcat, const bf16_t* __restrict__ WpB,
    const float* __restrict__ bproj, float* __restrict__ key)
{
    constexpr int K = KCAT_, N = NOUT_;
    constexpr int BM = 128, BN = 64, BK = 32, PAD = 8;
    __shared__ bf16_t As[BM][BK + PAD];
    __shared__ bf16_t Bs[BN][BK + PAD];

    const int tid  = threadIdx.x;
    const int wid  = tid >> 5;
    const int lane = tid & 31;
    const int half = lane >> 4;       // lane half selects K sub-chunks
    const int lrow = lane & 15;
    const int waveM = wid >> 1;       // 0..3 (32 rows each)
    const int waveN = wid & 1;        // 0..1 (32 cols each)
    const int m0 = blockIdx.y * BM;
    const int n0 = blockIdx.x * BN;

    v8f acc[2][2];
    #pragma unroll
    for (int a = 0; a < 2; ++a)
        #pragma unroll
        for (int b = 0; b < 2; ++b)
            acc[a][b] = (v8f){0.f,0.f,0.f,0.f,0.f,0.f,0.f,0.f};

    const int arow = tid >> 1, akh = (tid & 1) * 16;     // A: 32B/thread
    const int brow = tid >> 2, bkh = (tid & 3) * 8;      // B: 16B/thread

    for (int k0 = 0; k0 < K; k0 += BK) {
        const bf16_t* ga = Xcat + (size_t)(m0 + arow) * K + k0 + akh;
        const bf16_t* gb = WpB  + (size_t)(n0 + brow) * K + k0 + bkh;
#if HAVE_ASYNC_LDS
        ASYNC_CP16(ga, &As[arow][akh], 0);     // async Global->LDS, 16B/lane
        ASYNC_CP16(ga, &As[arow][akh], 16);
        ASYNC_CP16(gb, &Bs[brow][bkh], 0);
#else
        {
            const uint4* s = reinterpret_cast<const uint4*>(ga);
            uint4* d = reinterpret_cast<uint4*>(&As[arow][akh]);
            d[0] = s[0]; d[1] = s[1];
        }
        {
            const uint4* s = reinterpret_cast<const uint4*>(gb);
            uint4* d = reinterpret_cast<uint4*>(&Bs[brow][bkh]);
            d[0] = s[0];
        }
#endif
        if (k0 + BK < K) {  // prefetch next K tile -> global_prefetch_b8
            __builtin_prefetch(ga + BK, 0, 3);
            __builtin_prefetch(gb + BK, 0, 3);
        }
#if HAVE_ASYNC_LDS
        __builtin_amdgcn_s_wait_asynccnt(0);
#endif
        __syncthreads();

        v16bf afr[2], bfr[2];
        #pragma unroll
        for (int a = 0; a < 2; ++a) {
            int r = waveM * 32 + a * 16 + lrow;
            v8bf lo = *reinterpret_cast<const v8bf*>(&As[r][half * 8]);
            v8bf hi = *reinterpret_cast<const v8bf*>(&As[r][16 + half * 8]);
            afr[a] = join8(lo, hi);
        }
        #pragma unroll
        for (int b = 0; b < 2; ++b) {
            int c = waveN * 32 + b * 16 + lrow;
            v8bf lo = *reinterpret_cast<const v8bf*>(&Bs[c][half * 8]);
            v8bf hi = *reinterpret_cast<const v8bf*>(&Bs[c][16 + half * 8]);
            bfr[b] = join8(lo, hi);
        }
        #pragma unroll
        for (int a = 0; a < 2; ++a)
            #pragma unroll
            for (int b = 0; b < 2; ++b)
                acc[a][b] = __builtin_amdgcn_wmma_f32_16x16x32_bf16(
                    false, afr[a], false, bfr[b], (short)0, acc[a][b], false, false);
        __syncthreads();
    }

    // epilogue: D element (vgpr r, lane) -> M = r + 8*half, N = lrow
    #pragma unroll
    for (int a = 0; a < 2; ++a) {
        #pragma unroll
        for (int b = 0; b < 2; ++b) {
            int col = n0 + waveN * 32 + b * 16 + lrow;
            float bias = bproj[col];
            #pragma unroll
            for (int r = 0; r < 8; ++r) {
                int row = m0 + waveM * 32 + a * 16 + r + 8 * half;
                key[(size_t)row * N + col] = acc[a][b][r] + bias;
            }
        }
    }
}

// ---------------------------------------------------------------------------
// Attention: per batch row — logits, gated softmax, sel_h/sel_c.
// Fuses the rolling-shift writes of memory_h_new[:,1:] / memory_c_new[:,1:]
// so memory_h/c are swept exactly once. memory_h slab staged in LDS (60 KB).
// ---------------------------------------------------------------------------
__global__ __launch_bounds__(256) void attn_kernel(
    const float* __restrict__ key,    const float* __restrict__ memory_h,
    const float* __restrict__ memory_c, const float* __restrict__ gate_time,
    const float* __restrict__ rmask,
    float* __restrict__ outMH, float* __restrict__ outMC, float* __restrict__ outATT,
    float* __restrict__ selc,  bf16_t* __restrict__ A2)
{
    __shared__ float sh[NSLOTS_ * NOUT_];   // 61440 B
    __shared__ float slog[16];
    __shared__ float satt[16];

    const int b   = blockIdx.x;
    const int tid = threadIdx.x;
    const float* mhb = memory_h + (size_t)b * NSLOTS_ * NOUT_;
    const float* mcb = memory_c + (size_t)b * NSLOTS_ * NOUT_;

    if (tid < NSLOTS_) slog[tid] = 0.0f;

    // stage h slab + fused shift write of memory_h_new[:,1:]
    for (int idx = tid; idx < NSLOTS_ * NOUT_; idx += 256) {
        float hv = mhb[idx];
        sh[idx] = hv;
        int s = idx >> 10, d = idx & 1023;
        if (s < NSLOTS_ - 1)
            outMH[(size_t)b * NSLOTS_ * NOUT_ + (size_t)(s + 1) * NOUT_ + d] = hv;
    }
    __syncthreads();

    // logits[s] = dot(memory_h[b,s,:], key[b,:]) — register partials + ds_add_f32
    float part[NSLOTS_];
    #pragma unroll
    for (int s = 0; s < NSLOTS_; ++s) part[s] = 0.0f;
    #pragma unroll
    for (int j = 0; j < 4; ++j) {
        int d = tid + j * 256;
        float kv = key[(size_t)b * NOUT_ + d];
        #pragma unroll
        for (int s = 0; s < NSLOTS_; ++s) part[s] += sh[s * NOUT_ + d] * kv;
    }
    #pragma unroll
    for (int s = 0; s < NSLOTS_; ++s) atomicAdd(&slog[s], part[s]);
    __syncthreads();

    // gated softmax over 15 slots (serial — trivial cost)
    if (tid == 0) {
        const float scale = 0.03125f;   // 1/sqrt(1024)
        float mx = -1e30f;
        #pragma unroll
        for (int s = 0; s < NSLOTS_; ++s) {
            float l = slog[s] * scale;
            mx = fmaxf(mx, l);
        }
        float sum = 0.0f;
        #pragma unroll
        for (int s = 0; s < NSLOTS_; ++s) {
            float e = __expf(slog[s] * scale - mx) * gate_time[(size_t)b * NSLOTS_ + s];
            satt[s] = e;
            sum += e;
        }
        float inv = 1.0f / sum;
        #pragma unroll
        for (int s = 0; s < NSLOTS_; ++s) {
            satt[s] *= inv;
            outATT[(size_t)b * NSLOTS_ + s] = satt[s];
        }
    }
    __syncthreads();

    // sel_h/sel_c; fused shift of memory_c_new[:,1:]; emit bf16 GEMM2 operand
    #pragma unroll
    for (int j = 0; j < 4; ++j) {
        int d = tid + j * 256;
        float sH = 0.0f, sC = 0.0f;
        #pragma unroll
        for (int s = 0; s < NSLOTS_; ++s) {
            float a  = satt[s];
            sH += a * sh[s * NOUT_ + d];
            float cv = mcb[s * NOUT_ + d];
            sC += a * cv;
            if (s < NSLOTS_ - 1)
                outMC[(size_t)b * NSLOTS_ * NOUT_ + (size_t)(s + 1) * NOUT_ + d] = cv;
        }
        selc[(size_t)b * NOUT_ + d] = sC;
        A2[(size_t)b * KCAT_ + NINP_ + d] =
            (bf16_t)(sH * rmask[(size_t)b * NOUT_ + d]);
    }
}

// ---------------------------------------------------------------------------
// GEMM2 + LSTM: gates_g = A2[B,2048] @ W2[g][1024,2048]^T + bias2[g]
// Block tile 128x32, 8 waves (4x2): each wave owns 2 M-tiles x 4 gates for
// the SAME 16-col slice -> 8 WMMA / K-step, LSTM epilogue register-resident.
// ---------------------------------------------------------------------------
__global__ __launch_bounds__(256) void gemm_lstm_kernel(
    const bf16_t* __restrict__ A2, const bf16_t* __restrict__ W2,
    const float* __restrict__ bias2, const float* __restrict__ selc,
    float* __restrict__ outH, float* __restrict__ outMH, float* __restrict__ outMC)
{
    constexpr int K = KCAT_, N = NOUT_;
    constexpr int BM = 128, BN = 32, BK = 32, PAD = 8;
    __shared__ bf16_t As[BM][BK + PAD];
    __shared__ bf16_t Bs[4][BN][BK + PAD];

    const int tid  = threadIdx.x;
    const int wid  = tid >> 5;
    const int lane = tid & 31;
    const int half = lane >> 4;
    const int lrow = lane & 15;
    const int waveM = wid >> 1;       // 0..3 (32 rows each, 2 M-tiles)
    const int waveN = wid & 1;        // 0..1 (16 cols each)
    const int m0 = blockIdx.y * BM;
    const int n0 = blockIdx.x * BN;

    v8f acc[2][4];
    #pragma unroll
    for (int a = 0; a < 2; ++a)
        #pragma unroll
        for (int g = 0; g < 4; ++g)
            acc[a][g] = (v8f){0.f,0.f,0.f,0.f,0.f,0.f,0.f,0.f};

    const int arow = tid >> 1, akh = (tid & 1) * 16;                       // A: 32B/thread
    const int bg = tid >> 6, brow = (tid & 63) >> 1, bkh = (tid & 1) * 16; // B: 32B/thread

    for (int k0 = 0; k0 < K; k0 += BK) {
        const bf16_t* ga = A2 + (size_t)(m0 + arow) * K + k0 + akh;
        const bf16_t* gb = W2 + ((size_t)bg * NOUT_ + n0 + brow) * K + k0 + bkh;
#if HAVE_ASYNC_LDS
        ASYNC_CP16(ga, &As[arow][akh], 0);
        ASYNC_CP16(ga, &As[arow][akh], 16);
        ASYNC_CP16(gb, &Bs[bg][brow][bkh], 0);
        ASYNC_CP16(gb, &Bs[bg][brow][bkh], 16);
#else
        {
            const uint4* s = reinterpret_cast<const uint4*>(ga);
            uint4* d = reinterpret_cast<uint4*>(&As[arow][akh]);
            d[0] = s[0]; d[1] = s[1];
        }
        {
            const uint4* s = reinterpret_cast<const uint4*>(gb);
            uint4* d = reinterpret_cast<uint4*>(&Bs[bg][brow][bkh]);
            d[0] = s[0]; d[1] = s[1];
        }
#endif
        if (k0 + BK < K) {
            __builtin_prefetch(ga + BK, 0, 3);
            __builtin_prefetch(gb + BK, 0, 3);
        }
#if HAVE_ASYNC_LDS
        __builtin_amdgcn_s_wait_asynccnt(0);
#endif
        __syncthreads();

        v16bf afr[2];
        #pragma unroll
        for (int a = 0; a < 2; ++a) {
            int r = waveM * 32 + a * 16 + lrow;
            v8bf lo = *reinterpret_cast<const v8bf*>(&As[r][half * 8]);
            v8bf hi = *reinterpret_cast<const v8bf*>(&As[r][16 + half * 8]);
            afr[a] = join8(lo, hi);
        }
        #pragma unroll
        for (int g = 0; g < 4; ++g) {
            int c = waveN * 16 + lrow;
            v8bf lo = *reinterpret_cast<const v8bf*>(&Bs[g][c][half * 8]);
            v8bf hi = *reinterpret_cast<const v8bf*>(&Bs[g][c][16 + half * 8]);
            v16bf bfr = join8(lo, hi);
            #pragma unroll
            for (int a = 0; a < 2; ++a)
                acc[a][g] = __builtin_amdgcn_wmma_f32_16x16x32_bf16(
                    false, afr[a], false, bfr, (short)0, acc[a][g], false, false);
        }
        __syncthreads();
    }

    // LSTM epilogue — i/f/g/o for (row,col) all live in registers
    const int col = n0 + waveN * 16 + lrow;
    const float bi  = bias2[0 * NOUT_ + col];
    const float bf  = bias2[1 * NOUT_ + col];
    const float bg_ = bias2[2 * NOUT_ + col];
    const float bo  = bias2[3 * NOUT_ + col];
    #pragma unroll
    for (int a = 0; a < 2; ++a) {
        #pragma unroll
        for (int r = 0; r < 8; ++r) {
            int row = m0 + waveM * 32 + a * 16 + r + 8 * half;
            float iv = sigmoidf_(acc[a][0][r] + bi);
            float fv = sigmoidf_(acc[a][1][r] + bf);
            float gv = tanhf(acc[a][2][r] + bg_);
            float ov = sigmoidf_(acc[a][3][r] + bo);
            float cprev = selc[(size_t)row * N + col];
            float c = fv * cprev + iv * gv;
            float h = ov * tanhf(c);
            outH[(size_t)row * N + col] = h;
            outMH[(size_t)row * NSLOTS_ * NOUT_ + col] = h;   // slot 0
            outMC[(size_t)row * NSLOTS_ * NOUT_ + col] = c;   // slot 0
        }
    }
}

// ---------------------------------------------------------------------------
// Launch
// ---------------------------------------------------------------------------
extern "C" void kernel_launch(void* const* d_in, const int* in_sizes, int n_in,
                              void* d_out, int out_size, void* d_ws, size_t ws_size,
                              hipStream_t stream)
{
    (void)in_sizes; (void)n_in; (void)out_size; (void)ws_size;

    const float* input     = (const float*)d_in[0];
    const float* gate_time = (const float*)d_in[1];
    const float* rmask     = (const float*)d_in[2];
    const float* memory_h  = (const float*)d_in[3];
    const float* memory_c  = (const float*)d_in[4];
    const float* W_proj    = (const float*)d_in[5];
    const float* b_proj    = (const float*)d_in[6];
    const float* W_ih      = (const float*)d_in[7];
    const float* W_hh      = (const float*)d_in[8];
    const float* b_ih      = (const float*)d_in[9];
    const float* b_hh      = (const float*)d_in[10];

    // d_out layout: h_i | memory_h_new | memory_c_new | attention
    float* out    = (float*)d_out;
    float* outH   = out;
    float* outMH  = outH  + (size_t)B_ * NOUT_;
    float* outMC  = outMH + (size_t)B_ * NSLOTS_ * NOUT_;
    float* outATT = outMC + (size_t)B_ * NSLOTS_ * NOUT_;

    // workspace layout (~84 MB)
    char* ws = (char*)d_ws;
    bf16_t* Xcat  = (bf16_t*)ws;  ws += (size_t)B_ * KCAT_ * sizeof(bf16_t);        // 16 MB
    bf16_t* A2    = (bf16_t*)ws;  ws += (size_t)B_ * KCAT_ * sizeof(bf16_t);        // 16 MB
    bf16_t* WpB   = (bf16_t*)ws;  ws += (size_t)NOUT_ * KCAT_ * sizeof(bf16_t);     //  4 MB
    bf16_t* W2    = (bf16_t*)ws;  ws += (size_t)4 * NOUT_ * KCAT_ * sizeof(bf16_t); // 16 MB
    float*  bias2 = (float*)ws;   ws += (size_t)4 * NOUT_ * sizeof(float);          // 16 KB
    float*  key   = (float*)ws;   ws += (size_t)B_ * NOUT_ * sizeof(float);         // 16 MB
    float*  selc  = (float*)ws;   ws += (size_t)B_ * NOUT_ * sizeof(float);         // 16 MB

    // 1) pack f32 -> bf16
    pack_a_kernel  <<<(B_ * NINP_) / 256, 256, 0, stream>>>(input, memory_h, Xcat, A2);
    pack_wp_kernel <<<(NOUT_ * KCAT_) / 256, 256, 0, stream>>>(W_proj, WpB);
    pack_w2_kernel <<<(4 * NOUT_ * KCAT_) / 256, 256, 0, stream>>>(W_ih, W_hh, W2);
    pack_bias_kernel<<<(4 * NOUT_) / 256, 256, 0, stream>>>(b_ih, b_hh, bias2);

    // 2) key = [input | mh0] @ W_proj^T + b_proj
    {
        dim3 grid(NOUT_ / 64, B_ / 128);
        gemm_key_kernel<<<grid, 256, 0, stream>>>(Xcat, WpB, b_proj, key);
    }

    // 3) attention + sel + fused memory shift
    attn_kernel<<<B_, 256, 0, stream>>>(key, memory_h, memory_c, gate_time, rmask,
                                        outMH, outMC, outATT, selc, A2);

    // 4) gates GEMM + LSTM epilogue + slot-0 writes
    {
        dim3 grid(NOUT_ / 32, B_ / 128);
        gemm_lstm_kernel<<<grid, 256, 0, stream>>>(A2, W2, bias2, selc,
                                                   outH, outMH, outMC);
    }
}
